// NonLocal_HW_53772990546047
// MI455X (gfx1250) — matrix operand order repaired and checked by reading, not compiled, not run on hardware
//
#include <hip/hip_runtime.h>
#include <hip/hip_bf16.h>

#define BB 8
#define CC 256
#define CI 128
#define NN 4096

typedef __attribute__((ext_vector_type(16))) __bf16 v16bf;
typedef __attribute__((ext_vector_type(8)))  float  v8f;

union FragU { uint4 q[2]; v16bf v; };

static __device__ __forceinline__ unsigned short bf_bits(float f) {
    unsigned u = __builtin_bit_cast(unsigned, f);
    unsigned r = (u + 0x7FFFu + ((u >> 16) & 1u)) >> 16;   // RNE
    return (unsigned short)r;
}

// A-fragment (16x32 MxK, bf16): lane -> M=lane&15, K = {0..7}+8*half and {16..23}+8*half
// element (m,k) lives at base[m*stride + k0 + k]
static __device__ __forceinline__ v16bf loadA_bf16(const unsigned short* base, int stride, int k0, int lane) {
    int m = lane & 15, half = lane >> 4;
    const unsigned short* p = base + (size_t)m * stride + k0 + 8 * half;
    FragU u;
    u.q[0] = *(const uint4*)(p);
    u.q[1] = *(const uint4*)(p + 16);
    return u.v;
}
// B-fragment (32x16 KxN, bf16): lane -> N=lane&15, K = {0..7}+8*half and {16..23}+8*half
// element (k,n) lives at base[n*stride + k0 + k]
static __device__ __forceinline__ v16bf loadB_bf16(const unsigned short* base, int stride, int k0, int lane) {
    int n = lane & 15, half = lane >> 4;
    const unsigned short* p = base + (size_t)n * stride + k0 + 8 * half;
    FragU u;
    u.q[0] = *(const uint4*)(p);
    u.q[1] = *(const uint4*)(p + 16);
    return u.v;
}
static __device__ __forceinline__ v8f wmma_bf16(v16bf a, v16bf b, v8f c) {
    return __builtin_amdgcn_wmma_f32_16x16x32_bf16(false, a, false, b, (short)0, c, false, false);
}

// one-shot f32 -> bf16 elementwise conversion; n must be multiple of 4
__global__ __launch_bounds__(256) void cvt_bf16_kernel(const float* __restrict__ src,
                                                       unsigned short* __restrict__ dst, int n) {
    int i = (blockIdx.x * 256 + threadIdx.x) * 4;
    if (i >= n) return;
    float4 v = *(const float4*)(src + i);
    ushort4 o;
    o.x = bf_bits(v.x); o.y = bf_bits(v.y); o.z = bf_bits(v.z); o.w = bf_bits(v.w);
    *(ushort4*)(dst + i) = o;
}

// out[b][n][d] = sum_c w[d][c] * x[b][c][n] + bias[d]   (bf16 out, [B][N][CI]); x,w already bf16
__global__ __launch_bounds__(128) void proj_nd_kernel(const unsigned short* __restrict__ xbf,
                                                      const unsigned short* __restrict__ wb,
                                                      const float* __restrict__ bias,
                                                      unsigned short* __restrict__ out) {
    __shared__ unsigned short xT[64 * 40];   // [n=64][c=32], stride 40 (80B rows, 16B aligned)
    int tid = threadIdx.x, lane = tid & 31, wave = tid >> 5;
    int b = blockIdx.x >> 6;
    int n0 = (blockIdx.x & 63) << 6;
    const unsigned short* xb = xbf + (size_t)b * CC * NN;

    v8f acc[8] = {};
    for (int c0 = 0; c0 < CC; c0 += 32) {
        __syncthreads();
        for (int i = tid; i < 32 * 64; i += 128) {
            int c = i >> 6, n = i & 63;
            xT[n * 40 + c] = xb[(size_t)(c0 + c) * NN + n0 + n];
        }
        __syncthreads();
        v16bf a = loadA_bf16(xT + wave * 16 * 40, 40, 0, lane);
        v16bf bw[8];
#pragma unroll
        for (int dt = 0; dt < 8; ++dt)
            bw[dt] = loadB_bf16(wb + (size_t)(dt * 16) * CC, CC, c0, lane);
#pragma unroll
        for (int dt = 0; dt < 8; ++dt)
            acc[dt] = wmma_bf16(a, bw[dt], acc[dt]);
    }
    int half = lane >> 4, col = lane & 15;
    unsigned short* ob = out + ((size_t)b * NN + n0 + wave * 16) * CI;
#pragma unroll
    for (int dt = 0; dt < 8; ++dt) {
        float bia = bias[dt * 16 + col];
#pragma unroll
        for (int j = 0; j < 8; ++j) {
            int n = j + 8 * half;
            ob[(size_t)n * CI + dt * 16 + col] = bf_bits(acc[dt][j] + bia);
        }
    }
}

// out[b][d][n] = sum_c w[d][c] * x[b][c][n] + bias[d]   (bf16 out, [B][CI][N]); x,w already bf16
__global__ __launch_bounds__(128) void proj_dn_kernel(const unsigned short* __restrict__ xbf,
                                                      const unsigned short* __restrict__ wb,
                                                      const float* __restrict__ bias,
                                                      unsigned short* __restrict__ out) {
    __shared__ unsigned short xT[64 * 40];
    int tid = threadIdx.x, lane = tid & 31, wave = tid >> 5;
    int b = blockIdx.x >> 6;
    int n0 = (blockIdx.x & 63) << 6;
    const unsigned short* xb = xbf + (size_t)b * CC * NN;

    v8f acc[2][4] = {};
    for (int c0 = 0; c0 < CC; c0 += 32) {
        __syncthreads();
        for (int i = tid; i < 32 * 64; i += 128) {
            int c = i >> 6, n = i & 63;
            xT[n * 40 + c] = xb[(size_t)(c0 + c) * NN + n0 + n];
        }
        __syncthreads();
#pragma unroll
        for (int pass = 0; pass < 2; ++pass) {
            int d0 = pass * 64 + wave * 16;
            v16bf aw = loadA_bf16(wb + (size_t)d0 * CC, CC, c0, lane);
            v16bf bx[4];
#pragma unroll
            for (int nt = 0; nt < 4; ++nt)
                bx[nt] = loadB_bf16(xT + nt * 16 * 40, 40, 0, lane);
#pragma unroll
            for (int nt = 0; nt < 4; ++nt)
                acc[pass][nt] = wmma_bf16(aw, bx[nt], acc[pass][nt]);
        }
    }
    int half = lane >> 4, col = lane & 15;
#pragma unroll
    for (int pass = 0; pass < 2; ++pass) {
        int d0 = pass * 64 + wave * 16;
#pragma unroll
        for (int nt = 0; nt < 4; ++nt) {
#pragma unroll
            for (int j = 0; j < 8; ++j) {
                int d = d0 + j + 8 * half;
                int n = n0 + nt * 16 + col;
                out[((size_t)b * CI + d) * NN + n] = bf_bits(acc[pass][nt][j] + bias[d]);
            }
        }
    }
}

// Attention (shift-free softmax: scores are O(5), exp/sum safely in f32 range) + fused mask conv
// + residual. One wave per 16-query block. Row-sum of P via WMMA against all-ones B.
// Software-pipelined: QK WMMAs for tile m0+32 execute while tile m0's P does its LDS round trip.
__global__ __launch_bounds__(128) void attn_kernel(const unsigned short* __restrict__ theta,
                                                   const unsigned short* __restrict__ phi,
                                                   const unsigned short* __restrict__ g,
                                                   const unsigned short* __restrict__ wmask,
                                                   const float* __restrict__ b_mask,
                                                   const float* __restrict__ x,
                                                   float* __restrict__ out) {
    __shared__ unsigned short lds[4 * 2048];     // 4KB/wave: P tile [16][32], later Y tile [16][128]
    int tid = threadIdx.x, lane = tid & 31, wave = tid >> 5;
    unsigned short* myl = lds + wave * 2048;
    int half = lane >> 4, col = lane & 15;

    int qb = blockIdx.x * 4 + wave;              // 0..2047
    int b = qb >> 8;
    int nq0 = (qb & 255) << 4;

    const unsigned short* thb = theta + ((size_t)b * NN + nq0) * CI;
    const unsigned short* phb = phi + (size_t)b * NN * CI;
    const unsigned short* gb  = g + (size_t)b * CI * NN;

    v16bf aq[4];
#pragma unroll
    for (int kc = 0; kc < 4; ++kc) aq[kc] = loadA_bf16(thb, CI, kc * 32, lane);

    v16bf ones;
#pragma unroll
    for (int e = 0; e < 16; ++e) ones[e] = __builtin_bit_cast(__bf16, (unsigned short)0x3F80);

    v8f yacc[8] = {};
    v8f sumacc = {};                              // per-row sum of exp(scores), via P x ones

    // ---- prologue: scores for tile m0 = 0 ----
    v8f s0 = {}, s1 = {};
    {
        v16bf bq[8];
#pragma unroll
        for (int kc = 0; kc < 4; ++kc) {
            bq[kc]     = loadB_bf16(phb, CI, kc * 32, lane);
            bq[4 + kc] = loadB_bf16(phb + 16 * CI, CI, kc * 32, lane);
        }
#pragma unroll
        for (int kc = 0; kc < 4; ++kc) {
            s0 = wmma_bf16(aq[kc], bq[kc], s0);
            s1 = wmma_bf16(aq[kc], bq[4 + kc], s1);
        }
    }

    for (int m0 = 0; m0 < NN; m0 += 32) {
        bool hasNext = (m0 + 32) < NN;
        // (a) issue next QK fragment loads + current g fragment loads early
        v16bf bqn[8];
        if (hasNext) {
            const unsigned short* pbn = phb + (size_t)(m0 + 32) * CI;
#pragma unroll
            for (int kc = 0; kc < 4; ++kc) {
                bqn[kc]     = loadB_bf16(pbn, CI, kc * 32, lane);
                bqn[4 + kc] = loadB_bf16(pbn + 16 * CI, CI, kc * 32, lane);
            }
            __builtin_prefetch(pbn + 32 * CI, 0, 1);   // global_prefetch_b8 one more ahead
        }
        v16bf bg[8];
#pragma unroll
        for (int dt = 0; dt < 8; ++dt)
            bg[dt] = loadB_bf16(gb + (size_t)(dt * 16) * NN, NN, m0, lane);

        // (b) P = exp(S), store bf16 P tile [16][32] to wave-private LDS, reload as A-fragment
#pragma unroll
        for (int j = 0; j < 8; ++j) {
            int row = j + 8 * half;
            myl[row * 32 + col]      = bf_bits(__expf(s0[j]));
            myl[row * 32 + 16 + col] = bf_bits(__expf(s1[j]));
        }
        v16bf pa = loadA_bf16(myl, 32, 0, lane);

        // (c) QK WMMAs for the NEXT tile — independent, cover the LDS round-trip latency
        v8f t0 = {}, t1 = {};
        if (hasNext) {
#pragma unroll
            for (int kc = 0; kc < 4; ++kc) {
                t0 = wmma_bf16(aq[kc], bqn[kc], t0);
                t1 = wmma_bf16(aq[kc], bqn[4 + kc], t1);
            }
        }

        // (d) row-sum + PV WMMAs for current tile
        sumacc = wmma_bf16(pa, ones, sumacc);
#pragma unroll
        for (int dt = 0; dt < 8; ++dt)
            yacc[dt] = wmma_bf16(pa, bg[dt], yacc[dt]);

        s0 = t0; s1 = t1;
    }

    // normalize, stage y[16][128] bf16 into LDS
#pragma unroll
    for (int j = 0; j < 8; ++j) {
        float inv = 1.0f / sumacc[j];
        int row = j + 8 * half;
#pragma unroll
        for (int dt = 0; dt < 8; ++dt)
            myl[row * 128 + dt * 16 + col] = bf_bits(yacc[dt][j] * inv);
    }
    v16bf ya[4];
#pragma unroll
    for (int kc = 0; kc < 4; ++kc) ya[kc] = loadA_bf16(myl, 128, kc * 32, lane);

    // mask 1x1 conv: out[b][c][n] = sum_d w_mask[c][d]*y[n][d] + b_mask[c] + x[b][c][n]
    for (int ct = 0; ct < 16; ++ct) {
        v16bf bw[4];
#pragma unroll
        for (int kc = 0; kc < 4; ++kc)
            bw[kc] = loadB_bf16(wmask + (size_t)(ct * 16) * CI, CI, kc * 32, lane);
        v8f acc = {};
#pragma unroll
        for (int kc = 0; kc < 4; ++kc)
            acc = wmma_bf16(ya[kc], bw[kc], acc);
        int c = ct * 16 + col;
        float bia = b_mask[c];
        const float* xb = x + ((size_t)b * CC + c) * NN + nq0;
        float* ob = out + ((size_t)b * CC + c) * NN + nq0;
#pragma unroll
        for (int j = 0; j < 8; ++j) {
            int n = j + 8 * half;
            ob[n] = acc[j] + bia + xb[n];
        }
    }
}

extern "C" void kernel_launch(void* const* d_in, const int* in_sizes, int n_in,
                              void* d_out, int out_size, void* d_ws, size_t ws_size,
                              hipStream_t stream) {
    const float* x       = (const float*)d_in[0];
    const float* w_phi   = (const float*)d_in[1];
    const float* b_phi   = (const float*)d_in[2];
    const float* w_theta = (const float*)d_in[3];
    const float* b_theta = (const float*)d_in[4];
    const float* w_g     = (const float*)d_in[5];
    const float* b_g     = (const float*)d_in[6];
    const float* w_mask  = (const float*)d_in[7];
    const float* b_mask  = (const float*)d_in[8];
    float* out = (float*)d_out;

    const size_t xElem = (size_t)BB * CC * NN;              // 8.39M
    const size_t nElem = (size_t)BB * NN * CI;              // 4.19M each
    const int wElem = CI * CC;                              // 32768 per weight
    unsigned short* xbf   = (unsigned short*)d_ws;          // x bf16 [B][C][N]
    unsigned short* theta = xbf + xElem;                    // [B][N][CI] bf16
    unsigned short* phi   = theta + nElem;                  // [B][N][CI] bf16
    unsigned short* g     = phi + nElem;                    // [B][CI][N] bf16
    unsigned short* wtb   = g + nElem;                      // w_theta bf16 [CI][CC]
    unsigned short* wpb   = wtb + wElem;                    // w_phi   bf16 [CI][CC]
    unsigned short* wgb   = wpb + wElem;                    // w_g     bf16 [CI][CC]
    unsigned short* wmb   = wgb + wElem;                    // w_mask  bf16 [CC][CI]

    dim3 blk(128);
    cvt_bf16_kernel<<<dim3((unsigned)(xElem / 1024)), dim3(256), 0, stream>>>(x, xbf, (int)xElem);
    int cvtGrid = wElem / 1024;
    cvt_bf16_kernel<<<dim3(cvtGrid), dim3(256), 0, stream>>>(w_theta, wtb, wElem);
    cvt_bf16_kernel<<<dim3(cvtGrid), dim3(256), 0, stream>>>(w_phi,   wpb, wElem);
    cvt_bf16_kernel<<<dim3(cvtGrid), dim3(256), 0, stream>>>(w_g,     wgb, wElem);
    cvt_bf16_kernel<<<dim3(cvtGrid), dim3(256), 0, stream>>>(w_mask,  wmb, wElem);

    proj_nd_kernel<<<dim3(BB * (NN / 64)), blk, 0, stream>>>(xbf, wtb, b_theta, theta);
    proj_nd_kernel<<<dim3(BB * (NN / 64)), blk, 0, stream>>>(xbf, wpb, b_phi, phi);
    proj_dn_kernel<<<dim3(BB * (NN / 64)), blk, 0, stream>>>(xbf, wgb, b_g, g);
    attn_kernel<<<dim3(BB * (NN / 16) / 4), blk, 0, stream>>>(theta, phi, g, wmb, b_mask, x, out);
}
